// VLRAModel_20186346291790
// MI455X (gfx1250) — matrix operand order repaired and checked by reading, not compiled, not run on hardware
//
#include <hip/hip_runtime.h>
#include <hip/hip_bf16.h>

// ---------------------------------------------------------------------------
// Problem constants (from reference)
// ---------------------------------------------------------------------------
#define BB_    8
#define NN_    8192
#define HH_    512
#define WW_    512
#define CC_    192
#define HF_    128
#define WF_    128
#define STATE_ 256
#define VLR_   3
#define KPAD_  224         // 194/195 channels padded to 7*32
#define NTOT_  (BB_ * NN_) // 65536 points

// LDS row strides (bf16 units).  Both are multiples of 8 (16B alignment for
// b128 LDS ops) and chosen so 16 consecutive rows map to 16 distinct aligned
// 4-dword bank groups (232*2B=116dw: 52n mod 64 injective; 264*2B=132dw:
// 4n mod 64 injective) -> conflict-free B-fragment reads.
#define XSTR_  232
#define HSTR_  264

typedef __bf16 bf16_t;
typedef __attribute__((ext_vector_type(8)))  bf16_t v8bf;
typedef __attribute__((ext_vector_type(16))) bf16_t v16bf;
typedef __attribute__((ext_vector_type(8)))  float  v8f;

// ---------------------------------------------------------------------------
// Kernel 1: stride-4 "patchify" conv.  image (B,3,512,512) -> feat NHWC
// (B,128,128,192).  One block per output pixel; 4x4x3 patch in LDS.
// ---------------------------------------------------------------------------
__global__ void conv_patchify(const float* __restrict__ img,
                              const float* __restrict__ bw,   // (192,3,4,4)
                              const float* __restrict__ bb,
                              float* __restrict__ feat)
{
    const int sp = blockIdx.x;            // b*16384 + oy*128 + ox
    const int ox = sp & 127;
    const int oy = (sp >> 7) & 127;
    const int b  = sp >> 14;

    __shared__ float patch[48];
    const int t = threadIdx.x;            // 0..191
    if (t < 48) {
        const int ic = t >> 4;
        const int ry = (t >> 2) & 3;
        const int rx = t & 3;
        patch[t] = img[(((size_t)b * 3 + ic) * HH_ + (oy * 4 + ry)) * WW_ + (ox * 4 + rx)];
    }
    __syncthreads();

    float s = bb[t];
    const float* w = bw + (size_t)t * 48;
#pragma unroll
    for (int p = 0; p < 48; ++p) s += w[p] * patch[p];
    feat[(size_t)sp * CC_ + t] = s;
}

// ---------------------------------------------------------------------------
// Kernel 2: fp32 -> bf16 weight convert with K-padding.
// ---------------------------------------------------------------------------
__global__ void cvt_pad(const float* __restrict__ src, bf16_t* __restrict__ dst,
                        int rows, int kin, int kout)
{
    const int i = blockIdx.x * 256 + threadIdx.x;
    if (i >= rows * kout) return;
    const int r = i / kout;
    const int k = i - r * kout;
    dst[i] = (k < kin) ? (bf16_t)src[(size_t)r * kin + k] : (bf16_t)0.0f;
}

// ---------------------------------------------------------------------------
// Kernel 3: per-batch masked min of point coords (can_poly mins).
// ---------------------------------------------------------------------------
__global__ void minred(const float* __restrict__ py, const float* __restrict__ mask,
                       float* __restrict__ mins)
{
    __shared__ float sx[256], sy[256];
    const int b = blockIdx.x;
    float mx = 1e9f, my = 1e9f;
    for (int n = threadIdx.x; n < NN_; n += 256) {
        const int bn = b * NN_ + n;
        const float mk = mask[bn];
        const float vx = (mk > 0.0f) ? py[(size_t)bn * 2 + 0] : 1e9f;
        const float vy = (mk > 0.0f) ? py[(size_t)bn * 2 + 1] : 1e9f;
        mx = fminf(mx, vx);
        my = fminf(my, vy);
    }
    sx[threadIdx.x] = mx; sy[threadIdx.x] = my;
    __syncthreads();
    for (int s = 128; s > 0; s >>= 1) {
        if (threadIdx.x < s) {
            sx[threadIdx.x] = fminf(sx[threadIdx.x], sx[threadIdx.x + s]);
            sy[threadIdx.x] = fminf(sy[threadIdx.x], sy[threadIdx.x + s]);
        }
        __syncthreads();
    }
    if (threadIdx.x == 0) { mins[b * 2 + 0] = sx[0]; mins[b * 2 + 1] = sy[0]; }
}

// ---------------------------------------------------------------------------
// Kernel 4: FUSED per-iteration kernel.  One block = 64 points, 8 waves.
//   Phase A: bilinear sample + build bf16 X[64][KPAD] in LDS
//   Phase B: WMMA GEMM1  H1 = relu(W1 @ X + b1)   (H1 -> LDS, bf16)
//   Phase C: WMMA GEMM2  H2 = relu(W2 @ H1 + b2)  (H2 kept in registers);
//            head partials  o[row][n] += W3[row][m] * H2[m][n]  via ds_add_f32
//   Phase D: mode 0: py += (o + b3) * mask * STRIDE
//            mode 1: out = { py*DOWN , (o + b3) * mask }
// No intermediate global traffic; weights stream from L2.
// ---------------------------------------------------------------------------
__global__ void __launch_bounds__(256)
fused_iter(const float*  __restrict__ feat,   // NHWC
           float*        __restrict__ py,     // (B*N,2)
           const float*  __restrict__ mask,
           const float*  __restrict__ mins,   // (B,2)
           const bf16_t* __restrict__ W1,     // [256][KPAD_] bf16
           const float*  __restrict__ b1,     // [256]
           const bf16_t* __restrict__ W2,     // [256][256] bf16
           const float*  __restrict__ b2,     // [256]
           const float*  __restrict__ W3,     // (2,256) or (1,256) fp32
           const float*  __restrict__ b3,     // (2,) or (1,)
           int mode,                          // 0 = refine, 1 = classify
           float*        __restrict__ out)    // used in mode 1
{
    __shared__ __align__(16) bf16_t Xs[64 * XSTR_];   // 29.0 KB
    __shared__ __align__(16) bf16_t H1s[64 * HSTR_];  // 33.0 KB
    __shared__ float Oh[2][64];                       // head accumulators

    const int lane = threadIdx.x & 31;
    const int wave = threadIdx.x >> 5;
    const int lmod = lane & 15;
    const int lhi  = lane >> 4;
    const int bn0  = blockIdx.x * 64;

    // ---------------- Phase A: sample 64 points (wave-per-point) -----------
#pragma unroll
    for (int it = 0; it < 8; ++it) {
        const int p  = wave * 8 + it;
        const int bn = bn0 + p;
        const int b  = bn >> 13;
        const int n  = bn & (NN_ - 1);

        const float px = py[(size_t)bn * 2 + 0];
        const float pv = py[(size_t)bn * 2 + 1];
        const float m  = mask[bn];

        const float x = px * ((float)(WF_ - 1) / (float)WF_);
        const float y = pv * ((float)(HF_ - 1) / (float)HF_);
        const float x0 = floorf(x), y0 = floorf(y);
        const float fx = x - x0, fy = y - y0;
        const int x0i = (int)fminf(fmaxf(x0,        0.0f), (float)(WF_ - 1));
        const int x1i = (int)fminf(fmaxf(x0 + 1.0f, 0.0f), (float)(WF_ - 1));
        const int y0i = (int)fminf(fmaxf(y0,        0.0f), (float)(HF_ - 1));
        const int y1i = (int)fminf(fmaxf(y0 + 1.0f, 0.0f), (float)(HF_ - 1));
        const float w00 = (1.f - fy) * (1.f - fx);
        const float w01 = (1.f - fy) * fx;
        const float w10 = fy * (1.f - fx);
        const float w11 = fy * fx;

        const float* fb  = feat + (size_t)b * HF_ * WF_ * CC_;
        const float* p00 = fb + ((size_t)y0i * WF_ + x0i) * CC_;
        const float* p01 = fb + ((size_t)y0i * WF_ + x1i) * CC_;
        const float* p10 = fb + ((size_t)y1i * WF_ + x0i) * CC_;
        const float* p11 = fb + ((size_t)y1i * WF_ + x1i) * CC_;

        bf16_t* xr = Xs + (size_t)p * XSTR_;
#pragma unroll
        for (int i = 0; i < CC_ / 32; ++i) {
            const int c = lane + i * 32;
            const float v = w00 * p00[c] + w01 * p01[c] + w10 * p10[c] + w11 * p11[c];
            xr[c] = (bf16_t)(v * m);
        }
        if (mode == 0) {
            if (lane == 0) xr[192] = (bf16_t)((px - mins[b * 2 + 0]) * m);
            if (lane == 1) xr[193] = (bf16_t)((pv - mins[b * 2 + 1]) * m);
            if (lane >= 2) xr[192 + lane] = (bf16_t)0.0f;   // 194..223
        } else {
            if (lane == 0) xr[192] = (bf16_t)px;
            if (lane == 1) xr[193] = (bf16_t)pv;
            if (lane == 2) {
                const int npx = (n + 1) & (NN_ - 1);
                const int ppx = (n + NN_ - 1) & (NN_ - 1);
                const float ax = py[((size_t)b * NN_ + ppx) * 2 + 0] - px;
                const float ay = py[((size_t)b * NN_ + ppx) * 2 + 1] - pv;
                const float cx = py[((size_t)b * NN_ + npx) * 2 + 0] - px;
                const float cy = py[((size_t)b * NN_ + npx) * 2 + 1] - pv;
                const float dot = ax * cx + ay * cy;
                const float nrm = sqrtf(ax * ax + ay * ay) * sqrtf(cx * cx + cy * cy);
                const float cs  = fminf(fmaxf(dot / (nrm + 1e-8f), -1.0f), 1.0f);
                xr[194] = (bf16_t)(acosf(cs) * m);
            }
            if (lane >= 3) xr[192 + lane] = (bf16_t)0.0f;   // 195..223
        }
    }
    __syncthreads();

    // Wave tiling: 4 row-groups x 2 col-groups; wave = 4 row-tiles x 2 col-tiles.
    const int wr = wave & 3;
    const int wc = wave >> 2;
    const int mbase = wr * 64;
    const int ncol0 = wc * 32;      // local column base (0 or 32)

    // ---------------- Phase B: GEMM1 (K = KPAD_, B from Xs) ----------------
    {
        v8f acc[4][2];
#pragma unroll
        for (int rt = 0; rt < 4; ++rt)
#pragma unroll
            for (int ct = 0; ct < 2; ++ct)
#pragma unroll
                for (int r = 0; r < 8; ++r) acc[rt][ct][r] = 0.0f;

#pragma unroll
        for (int ks = 0; ks < KPAD_ / 32; ++ks) {
            const int k0 = ks * 32;
            v16bf afr[4];
#pragma unroll
            for (int rt = 0; rt < 4; ++rt) {
                const bf16_t* ap = W1 + (size_t)(mbase + rt * 16 + lmod) * KPAD_ + k0 + lhi * 16;
                v8bf lo = *(const v8bf*)(ap);
                v8bf hi = *(const v8bf*)(ap + 8);
                afr[rt] = __builtin_shufflevector(lo, hi, 0, 1, 2, 3, 4, 5, 6, 7,
                                                  8, 9, 10, 11, 12, 13, 14, 15);
            }
            v16bf bfr[2];
#pragma unroll
            for (int ct = 0; ct < 2; ++ct)
                bfr[ct] = *(const v16bf*)(Xs + (size_t)(ncol0 + ct * 16 + lmod) * XSTR_ + k0 + lhi * 16);
#pragma unroll
            for (int rt = 0; rt < 4; ++rt)
#pragma unroll
                for (int ct = 0; ct < 2; ++ct)
                    acc[rt][ct] = __builtin_amdgcn_wmma_f32_16x16x32_bf16(
                        false, afr[rt], false, bfr[ct], (short)0, acc[rt][ct],
                        false, false);
        }

        // bias + ReLU, pack 8 contiguous-M bf16 -> one 16B LDS store / tile
#pragma unroll
        for (int rt = 0; rt < 4; ++rt) {
            const int m0 = mbase + rt * 16 + lhi * 8;
#pragma unroll
            for (int ct = 0; ct < 2; ++ct) {
                const int p = ncol0 + ct * 16 + lmod;
                v8bf pk;
#pragma unroll
                for (int r = 0; r < 8; ++r) {
                    float v = acc[rt][ct][r] + b1[m0 + r];
                    pk[r] = (bf16_t)fmaxf(v, 0.0f);
                }
                *(v8bf*)(H1s + (size_t)p * HSTR_ + m0) = pk;
            }
        }
    }
    __syncthreads();

    // head accumulator init
    if (threadIdx.x < 128) Oh[threadIdx.x >> 6][threadIdx.x & 63] = 0.0f;
    __syncthreads();

    // ------------- Phase C: GEMM2 (K = 256, B from H1s) + head -------------
    {
        v8f acc[4][2];
#pragma unroll
        for (int rt = 0; rt < 4; ++rt)
#pragma unroll
            for (int ct = 0; ct < 2; ++ct)
#pragma unroll
                for (int r = 0; r < 8; ++r) acc[rt][ct][r] = 0.0f;

#pragma unroll
        for (int ks = 0; ks < STATE_ / 32; ++ks) {
            const int k0 = ks * 32;
            v16bf afr[4];
#pragma unroll
            for (int rt = 0; rt < 4; ++rt) {
                const bf16_t* ap = W2 + (size_t)(mbase + rt * 16 + lmod) * STATE_ + k0 + lhi * 16;
                v8bf lo = *(const v8bf*)(ap);
                v8bf hi = *(const v8bf*)(ap + 8);
                afr[rt] = __builtin_shufflevector(lo, hi, 0, 1, 2, 3, 4, 5, 6, 7,
                                                  8, 9, 10, 11, 12, 13, 14, 15);
            }
            v16bf bfr[2];
#pragma unroll
            for (int ct = 0; ct < 2; ++ct)
                bfr[ct] = *(const v16bf*)(H1s + (size_t)(ncol0 + ct * 16 + lmod) * HSTR_ + k0 + lhi * 16);
#pragma unroll
            for (int rt = 0; rt < 4; ++rt)
#pragma unroll
                for (int ct = 0; ct < 2; ++ct)
                    acc[rt][ct] = __builtin_amdgcn_wmma_f32_16x16x32_bf16(
                        false, afr[rt], false, bfr[ct], (short)0, acc[rt][ct],
                        false, false);
        }

        // bias + ReLU (H2 stays in registers) + head partial dot products.
        // Lane's 8 acc elements are M = m0..m0+7 at column n -> fold W3 here.
#pragma unroll
        for (int ct = 0; ct < 2; ++ct) {
            const int p = ncol0 + ct * 16 + lmod;   // local column
            float s0 = 0.0f, s1 = 0.0f;
#pragma unroll
            for (int rt = 0; rt < 4; ++rt) {
                const int m0 = mbase + rt * 16 + lhi * 8;
#pragma unroll
                for (int r = 0; r < 8; ++r) {
                    float h = fmaxf(acc[rt][ct][r] + b2[m0 + r], 0.0f);
                    s0 += W3[m0 + r] * h;
                    if (mode == 0) s1 += W3[STATE_ + m0 + r] * h;
                }
            }
            atomicAdd(&Oh[0][p], s0);
            if (mode == 0) atomicAdd(&Oh[1][p], s1);
        }
    }
    __syncthreads();

    // ---------------- Phase D: finalize ------------------------------------
    if (threadIdx.x < 64) {
        const int p  = threadIdx.x;
        const int bn = bn0 + p;
        const float m = mask[bn];
        if (mode == 0) {
            // py += off * mask * STRIDE
            py[(size_t)bn * 2 + 0] += (Oh[0][p] + b3[0]) * m * 4.0f;
            py[(size_t)bn * 2 + 1] += (Oh[1][p] + b3[1]) * m * 4.0f;
        } else {
            out[(size_t)bn * 2 + 0] = py[(size_t)bn * 2 + 0] * 4.0f;  // DOWN
            out[(size_t)bn * 2 + 1] = py[(size_t)bn * 2 + 1] * 4.0f;
            out[(size_t)NTOT_ * 2 + bn] = (Oh[0][p] + b3[0]) * m;
        }
    }
}

// ---------------------------------------------------------------------------
// Host launcher
// ---------------------------------------------------------------------------
extern "C" void kernel_launch(void* const* d_in, const int* in_sizes, int n_in,
                              void* d_out, int out_size, void* d_ws, size_t ws_size,
                              hipStream_t stream)
{
    (void)in_sizes; (void)n_in; (void)out_size; (void)ws_size;

    const float* image = (const float*)d_in[0];
    const float* pred  = (const float*)d_in[1];
    const float* vmask = (const float*)d_in[2];
    const float* bw    = (const float*)d_in[3];
    const float* bb    = (const float*)d_in[4];
    const float* oW1   = (const float*)d_in[5];
    const float* ob1   = (const float*)d_in[6];
    const float* oW2   = (const float*)d_in[7];
    const float* ob2   = (const float*)d_in[8];
    const float* oW3   = (const float*)d_in[9];
    const float* ob3   = (const float*)d_in[10];
    const float* cW1   = (const float*)d_in[11];
    const float* cb1   = (const float*)d_in[12];
    const float* cW2   = (const float*)d_in[13];
    const float* cb2   = (const float*)d_in[14];
    const float* cW3   = (const float*)d_in[15];
    const float* cb3   = (const float*)d_in[16];
    float* out = (float*)d_out;

    // workspace carve-up (feat + points + converted weights only)
    char* ws = (char*)d_ws;
    size_t o = 0;
    auto take = [&](size_t bytes) { char* p = ws + o; o += (bytes + 255) & ~(size_t)255; return p; };

    float*  feat  = (float*) take((size_t)BB_ * HF_ * WF_ * CC_ * sizeof(float)); // 96 MB NHWC
    float*  py    = (float*) take((size_t)NTOT_ * 2 * sizeof(float));
    float*  mins  = (float*) take((size_t)BB_ * 2 * sizeof(float));
    bf16_t* W1bf  = (bf16_t*)take((size_t)VLR_ * STATE_ * KPAD_  * sizeof(bf16_t));
    bf16_t* W2bf  = (bf16_t*)take((size_t)VLR_ * STATE_ * STATE_ * sizeof(bf16_t));
    bf16_t* cW1bf = (bf16_t*)take((size_t)STATE_ * KPAD_  * sizeof(bf16_t));
    bf16_t* cW2bf = (bf16_t*)take((size_t)STATE_ * STATE_ * sizeof(bf16_t));

    // 1) weight convert / pad (bf16)
    {
        int n;
        n = VLR_ * STATE_ * KPAD_;
        cvt_pad<<<(n + 255) / 256, 256, 0, stream>>>(oW1, W1bf, VLR_ * STATE_, 194, KPAD_);
        n = VLR_ * STATE_ * STATE_;
        cvt_pad<<<(n + 255) / 256, 256, 0, stream>>>(oW2, W2bf, VLR_ * STATE_, STATE_, STATE_);
        n = STATE_ * KPAD_;
        cvt_pad<<<(n + 255) / 256, 256, 0, stream>>>(cW1, cW1bf, STATE_, 195, KPAD_);
        n = STATE_ * STATE_;
        cvt_pad<<<(n + 255) / 256, 256, 0, stream>>>(cW2, cW2bf, STATE_, STATE_, STATE_);
    }

    // 2) patchify conv -> feat (NHWC)
    conv_patchify<<<BB_ * HF_ * WF_, CC_, 0, stream>>>(image, bw, bb, feat);

    // 3) py <- pred_points
    hipMemcpyAsync(py, pred, (size_t)NTOT_ * 2 * sizeof(float),
                   hipMemcpyDeviceToDevice, stream);

    const int fuse_blocks = NTOT_ / 64;   // 1024

    // 4) refinement iterations (fully fused: sample+MLP+update, zero
    //    intermediate global traffic)
    for (int i = 0; i < VLR_; ++i) {
        minred<<<BB_, 256, 0, stream>>>(py, vmask, mins);
        fused_iter<<<fuse_blocks, 256, 0, stream>>>(
            feat, py, vmask, mins,
            W1bf + (size_t)i * STATE_ * KPAD_,  ob1 + i * STATE_,
            W2bf + (size_t)i * STATE_ * STATE_, ob2 + i * STATE_,
            oW3 + (size_t)i * 2 * STATE_,       ob3 + i * 2,
            0, out);
    }

    // 5) classifier pass + outputs (same fused kernel, mode 1)
    fused_iter<<<fuse_blocks, 256, 0, stream>>>(
        feat, py, vmask, mins, cW1bf, cb1, cW2bf, cb2, cW3, cb3, 1, out);
}
